// MuNetSNN_29300266893847
// MI455X (gfx1250) — matrix operand order repaired and checked by reading, not compiled, not run on hardware
//
#include <hip/hip_runtime.h>

typedef _Float16 v16h __attribute__((ext_vector_type(16)));
typedef _Float16 v8h  __attribute__((ext_vector_type(8)));
typedef _Float16 v2h  __attribute__((ext_vector_type(2)));
typedef float    v8f  __attribute__((ext_vector_type(8)));

namespace snn {
constexpr int kB      = 16384;
constexpr int kState  = 24;
constexpr int kAction = 4;
constexpr int kHidden = 128;
constexpr int kSeq    = 40;
constexpr float kDtMem = 0.1f;   // dt * tau_mem_inv
constexpr float kDtSyn = 0.2f;   // dt * tau_syn_inv
constexpr float kVth   = 1.0f;
constexpr float kScale = 50.0f;

constexpr int kWaves      = 4;
constexpr int kBlock      = kWaves * 32;
constexpr int kRowsPerWv  = 16;
constexpr int kRowsPerWg  = kWaves * kRowsPerWv;

// LDS layout, in _Float16 units. Weight fragments are pre-packed in a contiguous-K (same K order as load_a)
// wave32 WMMA B-layout (lane l holds column n=l%16, K=(l&16)+e for element e),
// so every B fragment is a single contiguous 32B read per lane.
constexpr int kXsStride = 72;    // 48 spike cols + zero-pad to 64 + bank-skew pad
constexpr int kZStride  = 136;   // 128 spike cols + bank-skew pad
constexpr int kRecOff = 0;                               // [4 kc][8 nt][32 ln][16 e]
constexpr int kInOff  = kRecOff + 4 * 8 * 32 * 16;       // 16384
constexpr int kOutOff = kInOff  + 2 * 8 * 32 * 16;       // +8192
constexpr int kXsOff  = kOutOff + 4 * 32 * 16;           // +2048
constexpr int kZbOff  = kXsOff  + kWaves * 16 * kXsStride;
constexpr int kLdsF16 = kZbOff  + kWaves * 2 * 16 * kZStride;
constexpr int kLdsBytes = kLdsF16 * 2;                   // ~97 KB, fits WGP 320 KB
} // namespace snn

static __device__ __forceinline__ v16h combine8(v8h lo, v8h hi) {
  return __builtin_shufflevector(lo, hi, 0, 1, 2, 3, 4, 5, 6, 7,
                                         8, 9, 10, 11, 12, 13, 14, 15);
}
static __device__ __forceinline__ v8f zero8() {
  v8f z;
#pragma unroll
  for (int r = 0; r < 8; ++r) z[r] = 0.0f;
  return z;
}
// A fragment (16x32 f16) from a row-major LDS tile: lane l -> row l%16,
// lanes 0-15 take K {0..15}, lanes 16-31 take K {16..31}.
static __device__ __forceinline__ v16h load_a(const _Float16* base, int stride,
                                              int row, int kbase, int koff) {
  const _Float16* p = base + row * stride + kbase + koff;
  return combine8(*(const v8h*)p, *(const v8h*)(p + 8));
}

__global__ __launch_bounds__(snn::kBlock) void
MuNetSNN_wmma_kernel(const float* __restrict__ x, const float* __restrict__ w_in,
                     const float* __restrict__ w_rec, const float* __restrict__ w_out,
                     float* __restrict__ out) {
  using namespace snn;
  extern __shared__ __align__(128) _Float16 smem[];
  const int tid  = threadIdx.x;
  const int lane = tid & 31;
  const int wave = tid >> 5;
  const int row_l = lane & 15;          // local batch row within the wave tile
  const int half  = lane >> 4;          // encoder: 0 = +channels, 1 = -channels
  const int hi8   = (lane & 16) >> 1;   // C/D frag: lanes 16-31 hold rows M+8
  const int n_l   = lane & 15;          // C/D frag column
  const int rowg0 = blockIdx.x * kRowsPerWg + wave * kRowsPerWv;

  // ---- one-time init: zero spike staging, pack weights into B-fragments ----
  {
    unsigned* zp = (unsigned*)(smem + kXsOff);
    const int words = (kLdsF16 - kXsOff) / 2;
    for (int i = tid; i < words; i += kBlock) zp[i] = 0u;
  }
  for (int e = tid; e < 4 * 8 * 32 * 16; e += kBlock) {   // W_rec^T
    int el = e & 15, ln = (e >> 4) & 31, frag = e >> 9;
    int kc = frag >> 3, nt = frag & 7;
    int k = kc * 32 + (ln & 16) + el;
    int n = nt * 16 + (ln & 15);
    smem[kRecOff + e] = (_Float16)w_rec[n * kHidden + k];
  }
  for (int e = tid; e < 2 * 8 * 32 * 16; e += kBlock) {   // W_in^T, K padded 48->64
    int el = e & 15, ln = (e >> 4) & 31, frag = e >> 9;
    int kc = frag >> 3, nt = frag & 7;
    int k = kc * 32 + (ln & 16) + el;
    int n = nt * 16 + (ln & 15);
    smem[kInOff + e] = (_Float16)((k < 2 * kState) ? w_in[n * (2 * kState) + k] : 0.0f);
  }
  for (int e = tid; e < 4 * 32 * 16; e += kBlock) {       // W_out^T, N padded 4->16
    int el = e & 15, ln = (e >> 4) & 31, kc = e >> 9;
    int k = kc * 32 + (ln & 16) + el;
    int n = ln & 15;
    smem[kOutOff + e] = (_Float16)((n < kAction) ? w_out[n * kHidden + k] : 0.0f);
  }

  // ---- constant-current encoder state (wave-private, register resident) ----
  v2h  ienc[12];
  float venc[24];
  {
    const float* xr = x + (size_t)(rowg0 + row_l) * kState;
#pragma unroll
    for (int f = 0; f < 24; f += 2) {
      float a0 = xr[f] * kScale, a1 = xr[f + 1] * kScale;
      if (half) { a0 = -a0; a1 = -a1; }
      a0 = a0 > 0.0f ? a0 : 0.0f;
      a1 = a1 > 0.0f ? a1 : 0.0f;
      ienc[f >> 1] = (v2h){(_Float16)a0, (_Float16)a1};
      venc[f] = 0.0f; venc[f + 1] = 0.0f;
    }
  }
  __syncthreads();   // the ONLY barrier: weights + zeroed buffers visible

  // ---- wave-private recurrent state, in WMMA C/D fragment layout ----
  v8f v_h[8], i_h[8];
#pragma unroll
  for (int nt = 0; nt < 8; ++nt) { v_h[nt] = zero8(); i_h[nt] = zero8(); }
  v8f li_v = zero8(), li_i = zero8(), vmax;
#pragma unroll
  for (int r = 0; r < 8; ++r) vmax[r] = -3.4e38f;

  _Float16* xsW = smem + kXsOff + wave * 16 * kXsStride;
  _Float16* zb0 = smem + kZbOff + wave * 2 * 16 * kZStride;
  _Float16* zb1 = zb0 + 16 * kZStride;

  for (int t = 0; t < kSeq; ++t) {
    _Float16* zbR = (t & 1) ? zb1 : zb0;   // previous step's spikes
    _Float16* zbW = (t & 1) ? zb0 : zb1;   // this step's spikes

    // 1) encoder LIF step -> input spikes x_t (f16 {0,1}) into xs tile
#pragma unroll
    for (int f = 0; f < 24; f += 2) {
      float i0 = (float)ienc[f >> 1][0], i1 = (float)ienc[f >> 1][1];
      float v0 = venc[f]     + kDtMem * (i0 - venc[f]);
      float v1 = venc[f + 1] + kDtMem * (i1 - venc[f + 1]);
      float s0 = v0 > kVth ? 1.0f : 0.0f;
      float s1 = v1 > kVth ? 1.0f : 0.0f;
      venc[f]     = v0 - s0 * v0;
      venc[f + 1] = v1 - s1 * v1;
      *(v2h*)(xsW + row_l * kXsStride + half * kState + f) =
          (v2h){(_Float16)s0, (_Float16)s1};
    }

    // 2) hidden LIF membrane update; emit z_new into zbW (D-layout -> row-major)
#pragma unroll
    for (int nt = 0; nt < 8; ++nt) {
#pragma unroll
      for (int r = 0; r < 8; ++r) {
        float vv = v_h[nt][r], ii = i_h[nt][r];
        float vd = vv + kDtMem * (ii - vv);
        float z  = vd > kVth ? 1.0f : 0.0f;
        v_h[nt][r] = vd - z * vd;
        zbW[(r + hi8) * kZStride + nt * 16 + n_l] = (_Float16)z;
      }
    }

    // 3) A fragments: z_prev (K=128) and x_t (K=48 zero-padded to 64)
    v16h zA[4], xA[2];
#pragma unroll
    for (int kc = 0; kc < 4; ++kc)
      zA[kc] = load_a(zbR, kZStride, row_l, kc * 32, half * 16);
#pragma unroll
    for (int kc = 0; kc < 2; ++kc)
      xA[kc] = load_a(xsW, kXsStride, row_l, kc * 32, half * 16);

    // 4) i_new = (1-dt*tau_syn)*i + x_t@W_in^T + z_prev@W_rec^T  (48 WMMAs)
#pragma unroll
    for (int nt = 0; nt < 8; ++nt) {
      v8f acc = i_h[nt] * (1.0f - kDtSyn);
#pragma unroll
      for (int kc = 0; kc < 4; ++kc) {
        v16h b = *(const v16h*)(smem + kRecOff + ((kc * 8 + nt) * 32 + lane) * 16);
        acc = __builtin_amdgcn_wmma_f32_16x16x32_f16(false, zA[kc], false, b,
                                                     (short)0, acc, false, false);
      }
#pragma unroll
      for (int kc = 0; kc < 2; ++kc) {
        v16h b = *(const v16h*)(smem + kInOff + ((kc * 8 + nt) * 32 + lane) * 16);
        acc = __builtin_amdgcn_wmma_f32_16x16x32_f16(false, xA[kc], false, b,
                                                     (short)0, acc, false, false);
      }
      i_h[nt] = acc;
    }

    // 5) readout: i_jump = li_i + z_new@W_out^T (N padded to 16; 4 WMMAs)
    asm volatile("" ::: "memory");   // keep z_new stores ordered before reloads
    v8f acc = li_i;
#pragma unroll
    for (int kc = 0; kc < 4; ++kc) {
      v16h a = load_a(zbW, kZStride, row_l, kc * 32, half * 16);
      v16h b = *(const v16h*)(smem + kOutOff + (kc * 32 + lane) * 16);
      acc = __builtin_amdgcn_wmma_f32_16x16x32_f16(false, a, false, b,
                                                   (short)0, acc, false, false);
    }
    li_v = li_v + kDtMem * (acc - li_v);
    li_i = acc * (1.0f - kDtSyn);
#pragma unroll
    for (int r = 0; r < 8; ++r) vmax[r] = fmaxf(vmax[r], li_v[r]);
  }

  // ---- epilogue: out = tanh(max_t li_v), columns 0..3 of the padded tile ----
  if (n_l < kAction) {
#pragma unroll
    for (int r = 0; r < 8; ++r) {
      int rowg = rowg0 + r + hi8;
      out[(size_t)rowg * kAction + n_l] = tanhf(vmax[r]);
    }
  }
}

extern "C" void kernel_launch(void* const* d_in, const int* in_sizes, int n_in,
                              void* d_out, int out_size, void* d_ws, size_t ws_size,
                              hipStream_t stream) {
  (void)in_sizes; (void)n_in; (void)out_size; (void)d_ws; (void)ws_size;
  const float* x     = (const float*)d_in[0];
  const float* w_in  = (const float*)d_in[1];
  const float* w_rec = (const float*)d_in[2];
  const float* w_out = (const float*)d_in[3];
  float* out = (float*)d_out;
  dim3 grid(snn::kB / snn::kRowsPerWg);   // 256 workgroups
  dim3 block(snn::kBlock);                // 128 threads = 4 wave32
  MuNetSNN_wmma_kernel<<<grid, block, snn::kLdsBytes, stream>>>(x, w_in, w_rec,
                                                                w_out, out);
}
